// PointNetPP_62947040690655
// MI455X (gfx1250) — matrix operand order repaired and checked by reading, not compiled, not run on hardware
//
#include <hip/hip_runtime.h>
#include <hip/hip_bf16.h>
#include <math.h>

// ---------------------------------------------------------------------------
// PointNet++ forward for MI455X (gfx1250, wave32).
// Pipeline: FPS -> ball-query(top-32)+max-pool with WMMA point-MLPs, then
// global max pool and a WMMA MLP head + log_softmax.
// All matrix math uses v_wmma_f32_16x16x32_f16 (f16 in, f32 accumulate).
// Requires ws_size >= ~24 MB.
// ---------------------------------------------------------------------------

typedef __attribute__((ext_vector_type(16))) _Float16 v16h;
typedef __attribute__((ext_vector_type(8)))  float    v8f;

#define NEG_BIG 3.0e38f

// ---------------- WMMA fragment helpers (ISA 7.12.2 layouts) ----------------

__device__ __forceinline__ v8f wmma_f16(v16h a, v16h b, v8f c) {
  return __builtin_amdgcn_wmma_f32_16x16x32_f16(
      /*neg_a=*/false, a, /*neg_b=*/false, b,
      /*c_mod=*/(short)0, c, /*reuse_a=*/false, /*reuse_b=*/false);
}

// A fragment: 16x32 f16, row-major source (f16) with row stride ld (halfs).
// lane L: M = L&15 ; K-base = 8*(L>=16); elems 0..7 -> K=base+0..7,
// elems 8..15 -> K=16+base+0..7 (packed pairs per VGPR).
__device__ __forceinline__ v16h load_A_frag(const _Float16* src, int ld) {
  int lane = threadIdx.x & 31;
  int m = lane & 15;
  int kb = (lane >> 4) << 3;
  const _Float16* row = src + (size_t)m * ld;
  v16h a;
#pragma unroll
  for (int v = 0; v < 4; ++v) {
    a[2 * v]     = row[kb + 2 * v];
    a[2 * v + 1] = row[kb + 2 * v + 1];
    a[8 + 2 * v]     = row[16 + kb + 2 * v];
    a[8 + 2 * v + 1] = row[16 + kb + 2 * v + 1];
  }
  return a;
}

// Same as load_A_frag but f32 (global) source, converted to f16.
__device__ __forceinline__ v16h load_A_frag_gf32(const float* src, int ld) {
  int lane = threadIdx.x & 31;
  int m = lane & 15;
  int kb = (lane >> 4) << 3;
  const float* row = src + (size_t)m * ld;
  v16h a;
#pragma unroll
  for (int v = 0; v < 4; ++v) {
    a[2 * v]     = (_Float16)row[kb + 2 * v];
    a[2 * v + 1] = (_Float16)row[kb + 2 * v + 1];
    a[8 + 2 * v]     = (_Float16)row[16 + kb + 2 * v];
    a[8 + 2 * v + 1] = (_Float16)row[16 + kb + 2 * v + 1];
  }
  return a;
}

// B fragment: 32x16 f16 from row-major [K][N] source (f16, LDS).
// lane L: N = L&15 ; K-base = 8*(L>=16); elems 0..7 -> K=base+0..7, elems 8..15 -> K=16+base+0..7.
__device__ __forceinline__ v16h load_B_frag(const _Float16* src, int ld) {
  int lane = threadIdx.x & 31;
  int n = lane & 15;
  int kb = (lane >> 4) << 3;
  v16h b;
#pragma unroll
  for (int e = 0; e < 8; ++e) {
    b[e] = src[(size_t)(kb + e) * ld + n];
    b[8 + e] = src[(size_t)(16 + kb + e) * ld + n];
  }
  return b;
}

// B fragment from f32 global row-major [K][ld], with column validity bound.
__device__ __forceinline__ v16h load_B_frag_g(const float* W, int ld, int nvalid) {
  int lane = threadIdx.x & 31;
  int n = lane & 15;
  int kb = (lane >> 4) << 3;
  v16h b;
#pragma unroll
  for (int e = 0; e < 8; ++e) {
    b[e] = (_Float16)((n < nvalid) ? W[(size_t)(kb + e) * ld + n] : 0.0f);
    b[8 + e] = (_Float16)((n < nvalid) ? W[(size_t)(16 + kb + e) * ld + n] : 0.0f);
  }
  return b;
}

// C/D: lane L -> N = L&15, rows M = mbase..mbase+7 with mbase = 8*(L>=16).
__device__ __forceinline__ void store_D_f32(float* dst, int ld, v8f c) {
  int lane = threadIdx.x & 31;
  int n = lane & 15;
  int mbase = (lane >> 4) << 3;
#pragma unroll
  for (int r = 0; r < 8; ++r) dst[(size_t)(mbase + r) * ld + n] = c[r];
}

// ---------------- wave32 argmax (ties -> lower index) -----------------------

__device__ __forceinline__ void wave_argmax32(float v, int i, float& ov, int& oi) {
#pragma unroll
  for (int off = 16; off; off >>= 1) {
    float v2 = __shfl_xor(v, off, 32);
    int   i2 = __shfl_xor(i, off, 32);
    if (v2 > v || (v2 == v && i2 < i)) { v = v2; i = i2; }
  }
  ov = v; oi = i;
}

// ---------------- Farthest point sampling ------------------------------------
// One block per batch. Coords in LDS, per-point min-dist in registers.
// Emits centers[b][m][3] = pos[idx[m]] with idx[0]=0 (matches jax.lax.scan ys).
template <int NPTS>
__global__ __launch_bounds__(256) void fps_kernel(const float* __restrict__ pos,
                                                  float* __restrict__ centers,
                                                  int M) {
  constexpr int PT = NPTS / 256;
  __shared__ float px[NPTS], py[NPTS], pz[NPTS];
  __shared__ float rv[8];
  __shared__ int   ri[8];
  __shared__ int   s_cur;

  const int t = threadIdx.x, b = blockIdx.x;
  const float* P = pos + (size_t)b * NPTS * 3;
  for (int i = t; i < NPTS; i += 256) {
    px[i] = P[i * 3 + 0]; py[i] = P[i * 3 + 1]; pz[i] = P[i * 3 + 2];
  }
  float dist[PT];
#pragma unroll
  for (int j = 0; j < PT; ++j) dist[j] = NEG_BIG;  // "inf" init
  __syncthreads();

  int cur = 0;
  const int lane = t & 31;
  for (int m = 0; m < M; ++m) {
    const float cx = px[cur], cy = py[cur], cz = pz[cur];
    if (t == 0) {
      float* o = centers + ((size_t)b * M + m) * 3;
      o[0] = cx; o[1] = cy; o[2] = cz;
    }
    // update min-dists over this thread's contiguous slice; local argmax
    float bv = -1.0f; int bi = t * PT;
#pragma unroll
    for (int j = 0; j < PT; ++j) {
      const int i = t * PT + j;
      const float dx = px[i] - cx, dy = py[i] - cy, dz = pz[i] - cz;
      const float d = dx * dx + dy * dy + dz * dz;
      dist[j] = fminf(dist[j], d);
      if (dist[j] > bv) { bv = dist[j]; bi = i; }  // strict > : first index wins
    }
    float wv; int wi;
    wave_argmax32(bv, bi, wv, wi);
    if (lane == 0) { rv[t >> 5] = wv; ri[t >> 5] = wi; }
    __syncthreads();
    if (t == 0) {
      float fv = rv[0]; int fi = ri[0];
#pragma unroll
      for (int w = 1; w < 8; ++w)
        if (rv[w] > fv || (rv[w] == fv && ri[w] < fi)) { fv = rv[w]; fi = ri[w]; }
      s_cur = fi;
    }
    __syncthreads();
    cur = s_cur;
  }
}

// ---------------- SA1 point MLP: pos(3) -> 64 -> 64 --------------------------
// 256 threads = 8 waves; wave owns a 16-row tile; layer2 via WMMA.
__global__ __launch_bounds__(256) void mlp_sa1_kernel(
    const float* __restrict__ pos,  // [R][3]
    const float* __restrict__ w1, const float* __restrict__ b1,   // 3x64
    const float* __restrict__ w2, const float* __restrict__ b2,   // 64x64
    float* __restrict__ h)          // [R][64]
{
  __shared__ _Float16 sW2[64 * 64];
  __shared__ float sW1[3 * 64], sB1[64], sB2[64];
  __shared__ float sPos[128 * 3];
  __shared__ _Float16 sHid[8][16 * 64];

  const int t = threadIdx.x, wave = t >> 5, lane = t & 31;
  for (int i = t; i < 64 * 64; i += 256) sW2[i] = (_Float16)w2[i];
  for (int i = t; i < 3 * 64; i += 256) sW1[i] = w1[i];
  if (t < 64) { sB1[t] = b1[t]; sB2[t] = b2[t]; }
  for (int i = t; i < 128 * 3; i += 256)
    sPos[i] = pos[(size_t)blockIdx.x * 128 * 3 + i];
  __syncthreads();

  // layer1 (K=3, scalar FMAs), relu, store f16 tile
  _Float16* hid = sHid[wave];
  for (int e = lane; e < 16 * 64; e += 32) {
    const int r = e >> 6, c = e & 63;
    const float* p = &sPos[(wave * 16 + r) * 3];
    const float v = p[0] * sW1[c] + p[1] * sW1[64 + c] + p[2] * sW1[128 + c] + sB1[c];
    hid[e] = (_Float16)fmaxf(v, 0.0f);
  }
  __syncthreads();

  // layer2: hid[16][64] @ w2[64][64] + b2  via 2 WMMA per 16-col tile
  const int n = lane & 15;
  const int row0 = blockIdx.x * 128 + wave * 16;
  const v16h a0 = load_A_frag(hid, 64);
  const v16h a1 = load_A_frag(hid + 32, 64);
  float* orow = h + (size_t)row0 * 64;
#pragma unroll
  for (int nt = 0; nt < 4; ++nt) {
    v8f c;
    const float bias = sB2[nt * 16 + n];
#pragma unroll
    for (int r = 0; r < 8; ++r) c[r] = bias;
    c = wmma_f16(a0, load_B_frag(sW2 + nt * 16, 64), c);
    c = wmma_f16(a1, load_B_frag(sW2 + 32 * 64 + nt * 16, 64), c);
    store_D_f32(orow + nt * 16, 64, c);
  }
}

// ---------------- SA2 point MLP: x(64) -> 128 -> 128 -------------------------
__global__ __launch_bounds__(256) void mlp_sa2_kernel(
    const float* __restrict__ x,   // [R][64]
    const float* __restrict__ w1, const float* __restrict__ b1,  // 64x128
    const float* __restrict__ w2, const float* __restrict__ b2,  // 128x128
    float* __restrict__ h)         // [R][128]
{
  __shared__ _Float16 sW1[64 * 128];     // 16 KB
  __shared__ _Float16 sW2[128 * 128];    // 32 KB
  __shared__ float sB1[128], sB2[128];
  __shared__ _Float16 sHid[8][16 * 128]; // 32 KB

  const int t = threadIdx.x, wave = t >> 5, lane = t & 31;
  for (int i = t; i < 64 * 128; i += 256) sW1[i] = (_Float16)w1[i];
  for (int i = t; i < 128 * 128; i += 256) sW2[i] = (_Float16)w2[i];
  if (t < 128) { sB1[t] = b1[t]; sB2[t] = b2[t]; }
  __syncthreads();

  const int row0 = blockIdx.x * 128 + wave * 16;
  const int n = lane & 15, mbase = (lane >> 4) << 3;

  // layer1: x[16][64] @ w1[64][128] + b1, relu -> sHid (f16)
  const v16h a0 = load_A_frag_gf32(x + (size_t)row0 * 64, 64);
  const v16h a1 = load_A_frag_gf32(x + (size_t)row0 * 64 + 32, 64);
  _Float16* hid = sHid[wave];
#pragma unroll
  for (int nt = 0; nt < 8; ++nt) {
    v8f c;
    const float bias = sB1[nt * 16 + n];
#pragma unroll
    for (int r = 0; r < 8; ++r) c[r] = bias;
    c = wmma_f16(a0, load_B_frag(sW1 + nt * 16, 128), c);
    c = wmma_f16(a1, load_B_frag(sW1 + 32 * 128 + nt * 16, 128), c);
#pragma unroll
    for (int r = 0; r < 8; ++r)
      hid[(mbase + r) * 128 + nt * 16 + n] = (_Float16)fmaxf(c[r], 0.0f);
  }
  __syncthreads();

  // layer2: hid[16][128] @ w2[128][128] + b2 -> global f32
  v16h ha[4];
#pragma unroll
  for (int kc = 0; kc < 4; ++kc) ha[kc] = load_A_frag(hid + kc * 32, 128);
  float* orow = h + (size_t)row0 * 128;
#pragma unroll
  for (int nt = 0; nt < 8; ++nt) {
    v8f c;
    const float bias = sB2[nt * 16 + n];
#pragma unroll
    for (int r = 0; r < 8; ++r) c[r] = bias;
#pragma unroll
    for (int kc = 0; kc < 4; ++kc)
      c = wmma_f16(ha[kc], load_B_frag(sW2 + (size_t)kc * 32 * 128 + nt * 16, 128), c);
    store_D_f32(orow + nt * 16, 128, c);
  }
}

// ---------------- Ball query (top-32 in radius) + feature max-pool -----------
// One wave32 per center; K=32 => one candidate slot per lane.
template <int C>
__global__ __launch_bounds__(256) void ballquery_max_kernel(
    const float* __restrict__ pts,   // [B][N][3]
    const float* __restrict__ ctrs,  // [B][M][3]
    const float* __restrict__ feat,  // [B][N][C]
    float* __restrict__ out,         // [B][M][C]
    int N, int M, float r2)
{
  const int lane = threadIdx.x & 31;
  const int gw = blockIdx.x * (blockDim.x >> 5) + (threadIdx.x >> 5);
  const int b = gw / M, m = gw % M;

  const float cx = ctrs[((size_t)b * M + m) * 3 + 0];
  const float cy = ctrs[((size_t)b * M + m) * 3 + 1];
  const float cz = ctrs[((size_t)b * M + m) * 3 + 2];
  const float* P = pts + (size_t)b * N * 3;

  float slot_d = NEG_BIG;  // per-lane slot (one of current top-32)
  int   slot_i = -1;

  for (int base = 0; base < N; base += 32) {
    const int pi = base + lane;
    const float dx = P[pi * 3 + 0] - cx;
    const float dy = P[pi * 3 + 1] - cy;
    const float dz = P[pi * 3 + 2] - cz;
    const float d2 = dx * dx + dy * dy + dz * dz;
    unsigned mask = (unsigned)__ballot(d2 <= r2);
    while (mask) {
      const int src = __builtin_ctz(mask);
      mask &= mask - 1;
      const float d = __shfl(d2, src, 32);
      float wv; int wl;
      wave_argmax32(slot_d, lane, wv, wl);   // current worst slot
      if (d < wv) {
        if (lane == wl) { slot_d = d; slot_i = base + src; }
      }
    }
  }

  // max-pool features over selected neighbors
  constexpr int CPL = C / 32;
  float acc[CPL];
#pragma unroll
  for (int j = 0; j < CPL; ++j) acc[j] = -NEG_BIG;
  for (int s = 0; s < 32; ++s) {
    const int idx = __shfl(slot_i, s, 32);
    if (idx < 0) continue;
    const float* f = feat + ((size_t)b * N + idx) * C;
#pragma unroll
    for (int j = 0; j < CPL; ++j) acc[j] = fmaxf(acc[j], f[lane + 32 * j]);
  }
  float* o = out + ((size_t)b * M + m) * C;
#pragma unroll
  for (int j = 0; j < CPL; ++j) o[lane + 32 * j] = acc[j];
}

// ---------------- Global max pool over M2 ------------------------------------
__global__ void gpool_kernel(const float* __restrict__ x2, float* __restrict__ g) {
  const int b = blockIdx.x, c = threadIdx.x;  // 128 threads
  float mx = -NEG_BIG;
  for (int m = 0; m < 512; ++m)
    mx = fmaxf(mx, x2[((size_t)b * 512 + m) * 128 + c]);
  g[b * 128 + c] = mx;
}

// ---------------- Head: 128->256(relu)->512, ->256(relu), ->10, log_softmax --
__device__ __forceinline__ void head_layer(const _Float16* inLds, int K,
                                           const float* __restrict__ W,
                                           const float* __restrict__ Bv,
                                           int Nout, _Float16* outLds,
                                           bool relu_flag) {
  const int t = threadIdx.x, wave = t >> 5, lane = t & 31;
  const int n = lane & 15, mbase = (lane >> 4) << 3;
  const int ntiles = Nout >> 4;
  for (int nt = wave; nt < ntiles; nt += 8) {
    v8f c;
    const float bias = Bv[nt * 16 + n];
#pragma unroll
    for (int r = 0; r < 8; ++r) c[r] = bias;
    for (int kc = 0; kc < (K >> 5); ++kc) {
      const v16h a = load_A_frag(inLds + kc * 32, K);
      const v16h bf = load_B_frag_g(W + (size_t)(kc * 32) * Nout + nt * 16, Nout, 16);
      c = wmma_f16(a, bf, c);
    }
#pragma unroll
    for (int r = 0; r < 8; ++r) {
      const float v = relu_flag ? fmaxf(c[r], 0.0f) : c[r];
      outLds[(mbase + r) * Nout + nt * 16 + n] = (_Float16)v;
    }
  }
}

__global__ __launch_bounds__(256) void head_kernel(
    const float* __restrict__ g,                                   // [8][128]
    const float* __restrict__ w3a, const float* __restrict__ b3a,  // 128x256
    const float* __restrict__ w3b, const float* __restrict__ b3b,  // 256x512
    const float* __restrict__ f1w, const float* __restrict__ f1b,  // 512x256
    const float* __restrict__ f2w, const float* __restrict__ f2b,  // 256x10
    float* __restrict__ out)                                       // [8][10]
{
  __shared__ _Float16 sA[16 * 128];
  __shared__ _Float16 sH1[16 * 256];
  __shared__ _Float16 sH2[16 * 512];
  __shared__ _Float16 sH3[16 * 256];
  __shared__ float sLog[16 * 16];

  const int t = threadIdx.x, wave = t >> 5, lane = t & 31;
  for (int i = t; i < 16 * 128; i += 256) {
    const int r = i >> 7, c = i & 127;
    sA[i] = (_Float16)((r < 8) ? g[r * 128 + c] : 0.0f);  // pad rows 8..15
  }
  __syncthreads();
  head_layer(sA, 128, w3a, b3a, 256, sH1, true);   __syncthreads();
  head_layer(sH1, 256, w3b, b3b, 512, sH2, false); __syncthreads();
  head_layer(sH2, 512, f1w, f1b, 256, sH3, true);  __syncthreads();

  // final 256 -> 10 (padded to one 16-col tile), wave 0 only
  if (wave == 0) {
    const int n = lane & 15, mbase = (lane >> 4) << 3;
    v8f c;
    const float bias = (n < 10) ? f2b[n] : 0.0f;
#pragma unroll
    for (int r = 0; r < 8; ++r) c[r] = bias;
#pragma unroll
    for (int kc = 0; kc < 8; ++kc) {
      const v16h a = load_A_frag(sH3 + kc * 32, 256);
      const v16h bf = load_B_frag_g(f2w + (size_t)(kc * 32) * 10, 10, 10);
      c = wmma_f16(a, bf, c);
    }
#pragma unroll
    for (int r = 0; r < 8; ++r) sLog[(mbase + r) * 16 + n] = c[r];
  }
  __syncthreads();

  if (t < 8) {
    float mx = -NEG_BIG;
    for (int j = 0; j < 10; ++j) mx = fmaxf(mx, sLog[t * 16 + j]);
    float s = 0.0f;
    for (int j = 0; j < 10; ++j) s += __expf(sLog[t * 16 + j] - mx);
    const float lse = mx + __logf(s);
    for (int j = 0; j < 10; ++j) out[t * 10 + j] = sLog[t * 16 + j] - lse;
  }
}

// ---------------------------------------------------------------------------

extern "C" void kernel_launch(void* const* d_in, const int* in_sizes, int n_in,
                              void* d_out, int out_size, void* d_ws, size_t ws_size,
                              hipStream_t stream) {
  (void)in_sizes; (void)n_in; (void)out_size; (void)ws_size;
  const float* pos  = (const float*)d_in[0];
  const float* w1a  = (const float*)d_in[1];
  const float* b1a  = (const float*)d_in[2];
  const float* w1b  = (const float*)d_in[3];
  const float* b1b  = (const float*)d_in[4];
  const float* w2a  = (const float*)d_in[5];
  const float* b2a  = (const float*)d_in[6];
  const float* w2b  = (const float*)d_in[7];
  const float* b2b  = (const float*)d_in[8];
  const float* w3a  = (const float*)d_in[9];
  const float* b3a  = (const float*)d_in[10];
  const float* w3b  = (const float*)d_in[11];
  const float* b3b  = (const float*)d_in[12];
  const float* f1w  = (const float*)d_in[13];
  const float* f1b  = (const float*)d_in[14];
  const float* f2w  = (const float*)d_in[15];
  const float* f2b  = (const float*)d_in[16];
  float* out = (float*)d_out;

  // workspace layout (floats); total 5,829,632 floats ~= 23.3 MB
  float* ws = (float*)d_ws;
  float* c1 = ws;                 // [8][2048][3]
  float* h1 = c1 + 49152;         // [8][4096][64]
  float* x1 = h1 + 2097152;       // [8][2048][64]
  float* c2 = x1 + 1048576;       // [8][512][3]
  float* h2 = c2 + 12288;         // [8][2048][128]
  float* x2 = h2 + 2097152;       // [8][512][128]
  float* gg = x2 + 524288;        // [8][128]

  // SA module 1
  fps_kernel<4096><<<8, 256, 0, stream>>>(pos, c1, 2048);
  mlp_sa1_kernel<<<256, 256, 0, stream>>>(pos, w1a, b1a, w1b, b1b, h1);
  ballquery_max_kernel<64><<<2048, 256, 0, stream>>>(pos, c1, h1, x1,
                                                     4096, 2048, 0.2f * 0.2f);
  // SA module 2
  fps_kernel<2048><<<8, 256, 0, stream>>>(c1, c2, 512);
  mlp_sa2_kernel<<<128, 256, 0, stream>>>(x1, w2a, b2a, w2b, b2b, h2);
  ballquery_max_kernel<128><<<512, 256, 0, stream>>>(c1, c2, h2, x2,
                                                     2048, 512, 0.4f * 0.4f);
  // global pool + head
  gpool_kernel<<<8, 128, 0, stream>>>(x2, gg);
  head_kernel<<<1, 256, 0, stream>>>(gg, w3a, b3a, w3b, b3b, f1w, f1b, f2w, f2b, out);
}
